// SimplifiedAttention_69080253989708
// MI455X (gfx1250) — matrix operand order repaired and checked by reading, not compiled, not run on hardware
//
#include <hip/hip_runtime.h>

typedef _Float16 v16h __attribute__((ext_vector_type(16)));
typedef _Float16 v8h  __attribute__((ext_vector_type(8)));
typedef float    v8f  __attribute__((ext_vector_type(8)));

#define BATCH 4
#define SEQ   4096
#define EMB   1024
#define EPSN  1e-12f
#define LDSK  40            // 32 k-halves + 8 pad (keeps 16B alignment, spreads banks)

union FragU { v16h v; v8h h[2]; };

__device__ __forceinline__ v16h frag_from(const _Float16* p0, const _Float16* p1) {
  FragU u;
  u.h[0] = *(const v8h*)p0;   // 16B LDS load
  u.h[1] = *(const v8h*)p1;   // 16B LDS load
  return u.v;
}

__device__ __forceinline__ void split_store(float x, _Float16* hi, _Float16* lo) {
  _Float16 h = (_Float16)x;
  *hi = h;
  *lo = (_Float16)(x - (float)h);
}

// ---------------------------------------------------------------------------
// 1) inverse L2 norms of q and k rows: one wave32 per row
// ---------------------------------------------------------------------------
__global__ __launch_bounds__(256) void norms_kernel(const float* __restrict__ q,
                                                    const float* __restrict__ k,
                                                    float* __restrict__ norms) {
  const int wid  = threadIdx.x >> 5;
  const int lane = threadIdx.x & 31;
  int gRow = blockIdx.x * 8 + wid;                 // 0 .. 2*BATCH*SEQ-1
  const float* src;
  float* dst;
  int r;
  if (gRow < BATCH * SEQ) { src = q; dst = norms;               r = gRow; }
  else                    { src = k; dst = norms + BATCH * SEQ; r = gRow - BATCH * SEQ; }
  const float* row = src + (size_t)r * EMB;
  float s = 0.f;
#pragma unroll
  for (int j = 0; j < 8; ++j) {
    float4 v = *reinterpret_cast<const float4*>(row + j * 128 + lane * 4);
    s += v.x * v.x + v.y * v.y + v.z * v.z + v.w * v.w;
  }
#pragma unroll
  for (int off = 16; off > 0; off >>= 1) s += __shfl_xor(s, off, 32);
  if (lane == 0) dst[r] = 1.0f / fmaxf(sqrtf(s), EPSN);
}

// ---------------------------------------------------------------------------
// 2) S = (rq*Q) @ (rk*K)^T, 128x128 tile per block, f16x3 split WMMA,
//    register-level software pipeline over the k-loop.
// ---------------------------------------------------------------------------
__global__ __launch_bounds__(256) void qk_gemm_kernel(const float* __restrict__ q,
                                                      const float* __restrict__ k,
                                                      const float* __restrict__ norms,
                                                      float* __restrict__ w) {
  __shared__ _Float16 sHiA[128 * LDSK], sLoA[128 * LDSK];
  __shared__ _Float16 sHiB[128 * LDSK], sLoB[128 * LDSK];

  const int tid  = threadIdx.x;
  const int lane = tid & 31, wid = tid >> 5;
  const int bat  = blockIdx.z;
  const int mBase = blockIdx.y * 128, nBase = blockIdx.x * 128;

  const float* qb = q + (size_t)bat * SEQ * EMB;
  const float* kb = k + (size_t)bat * SEQ * EMB;
  const float* rq = norms + (size_t)bat * SEQ;
  const float* rk = norms + (size_t)(BATCH + bat) * SEQ;
  float* wb = w + (size_t)bat * SEQ * SEQ;

  const int wm = (wid & 1) * 64, wn = (wid >> 1) * 32;
  const int hsel = (lane >= 16) ? 1 : 0;
  const int c0  = hsel * 8;       // A-frag k chunk start (ISA 16-bit A layout)
  const int kb0 = hsel * 16;      // unused; B frags use c0 (same k order as A)
  const int mr  = lane & 15;

  const int rp = tid >> 3;        // staging row within a 32-row pass
  const int kk = (tid & 7) * 4;   // staging k offset

  v8f acc[8];
#pragma unroll
  for (int t = 0; t < 8; ++t) acc[t] = (v8f){0.f, 0.f, 0.f, 0.f, 0.f, 0.f, 0.f, 0.f};

  // loop-invariant row scales + prefetch of k-step 0
  float  sa[4], sb[4];
  float4 qreg[4], kreg[4];
#pragma unroll
  for (int p = 0; p < 4; ++p) {
    const int r = p * 32 + rp;
    sa[p]   = rq[mBase + r];
    sb[p]   = rk[nBase + r];
    qreg[p] = *reinterpret_cast<const float4*>(qb + (size_t)(mBase + r) * EMB + kk);
    kreg[p] = *reinterpret_cast<const float4*>(kb + (size_t)(nBase + r) * EMB + kk);
  }

  for (int ks = 0; ks < EMB / 32; ++ks) {
    // convert + store the prefetched tile into LDS (hi/lo f16 planes)
#pragma unroll
    for (int p = 0; p < 4; ++p) {
      const int r = p * 32 + rp;
      float xs[4] = {qreg[p].x * sa[p], qreg[p].y * sa[p], qreg[p].z * sa[p], qreg[p].w * sa[p]};
      float ys[4] = {kreg[p].x * sb[p], kreg[p].y * sb[p], kreg[p].z * sb[p], kreg[p].w * sb[p]};
#pragma unroll
      for (int e = 0; e < 4; ++e) {
        split_store(xs[e], &sHiA[r * LDSK + kk + e], &sLoA[r * LDSK + kk + e]);
        split_store(ys[e], &sHiB[r * LDSK + kk + e], &sLoB[r * LDSK + kk + e]);
      }
    }
    __syncthreads();

    // issue next k-step's global loads; latency hidden behind the WMMA block
    if (ks + 1 < EMB / 32) {
      const int k0n = (ks + 1) * 32;
#pragma unroll
      for (int p = 0; p < 4; ++p) {
        const int r = p * 32 + rp;
        qreg[p] = *reinterpret_cast<const float4*>(qb + (size_t)(mBase + r) * EMB + k0n + kk);
        kreg[p] = *reinterpret_cast<const float4*>(kb + (size_t)(nBase + r) * EMB + k0n + kk);
      }
    }

    v16h aHi[4], aLo[4], bHi[2], bLo[2];
#pragma unroll
    for (int mi = 0; mi < 4; ++mi) {
      const int base = (wm + mi * 16 + mr) * LDSK;
      aHi[mi] = frag_from(&sHiA[base + c0], &sHiA[base + c0 + 16]);
      aLo[mi] = frag_from(&sLoA[base + c0], &sLoA[base + c0 + 16]);
    }
#pragma unroll
    for (int ni = 0; ni < 2; ++ni) {
      const int base = (wn + ni * 16 + mr) * LDSK;
      bHi[ni] = frag_from(&sHiB[base + c0], &sHiB[base + c0 + 16]);
      bLo[ni] = frag_from(&sLoB[base + c0], &sLoB[base + c0 + 16]);
    }
#pragma unroll
    for (int mi = 0; mi < 4; ++mi)
#pragma unroll
      for (int ni = 0; ni < 2; ++ni) {
        const int t = mi * 2 + ni;
        acc[t] = __builtin_amdgcn_wmma_f32_16x16x32_f16(false, aHi[mi], false, bHi[ni], (short)0, acc[t], false, false);
        acc[t] = __builtin_amdgcn_wmma_f32_16x16x32_f16(false, aHi[mi], false, bLo[ni], (short)0, acc[t], false, false);
        acc[t] = __builtin_amdgcn_wmma_f32_16x16x32_f16(false, aLo[mi], false, bHi[ni], (short)0, acc[t], false, false);
      }
    __syncthreads();
  }

  // write S tile (C/D layout: VGPR r -> M = r + 8*hsel, N = lane&15)
#pragma unroll
  for (int mi = 0; mi < 4; ++mi)
#pragma unroll
    for (int ni = 0; ni < 2; ++ni) {
      const int t = mi * 2 + ni;
      const int n = nBase + wn + ni * 16 + mr;
#pragma unroll
      for (int r = 0; r < 8; ++r) {
        const int m = mBase + wm + mi * 16 + hsel * 8 + r;
        wb[(size_t)m * SEQ + n] = acc[t][r];
      }
    }
}

// ---------------------------------------------------------------------------
// 3) O = S @ V, same tiling; V transposed into LDS during staging,
//    register-level software pipeline over the k-loop.
// ---------------------------------------------------------------------------
__global__ __launch_bounds__(256) void sv_gemm_kernel(const float* __restrict__ w,
                                                      const float* __restrict__ v,
                                                      float* __restrict__ o) {
  __shared__ _Float16 sHiA[128 * LDSK], sLoA[128 * LDSK];
  __shared__ _Float16 sHiB[128 * LDSK], sLoB[128 * LDSK];

  const int tid  = threadIdx.x;
  const int lane = tid & 31, wid = tid >> 5;
  const int bat  = blockIdx.z;
  const int mBase = blockIdx.y * 128, nBase = blockIdx.x * 128;

  const float* wbuf = w + (size_t)bat * SEQ * SEQ;
  const float* vb   = v + (size_t)bat * SEQ * EMB;
  float* ob = o + (size_t)bat * SEQ * EMB;

  const int wm = (wid & 1) * 64, wn = (wid >> 1) * 32;
  const int hsel = (lane >= 16) ? 1 : 0;
  const int c0  = hsel * 8;
  const int kb0 = hsel * 16;
  const int mr  = lane & 15;

  const int rp = tid >> 3;        // A staging row within a 32-row pass
  const int kk = (tid & 7) * 4;   // A staging k offset
  const int nn = tid & 127;       // B staging column
  const int kt = tid >> 7;        // B staging k sub-offset (0/1)

  v8f acc[8];
#pragma unroll
  for (int t = 0; t < 8; ++t) acc[t] = (v8f){0.f, 0.f, 0.f, 0.f, 0.f, 0.f, 0.f, 0.f};

  // prefetch k-step 0
  float4 areg[4];
  float  breg[16];
#pragma unroll
  for (int p = 0; p < 4; ++p)
    areg[p] = *reinterpret_cast<const float4*>(wbuf + (size_t)(mBase + p * 32 + rp) * SEQ + kk);
#pragma unroll
  for (int p = 0; p < 16; ++p)
    breg[p] = vb[(size_t)(p * 2 + kt) * EMB + nBase + nn];

  for (int ks = 0; ks < SEQ / 32; ++ks) {
    // convert + store prefetched tiles: A row-major, B transposed to [n][k]
#pragma unroll
    for (int p = 0; p < 4; ++p) {
      const int r = p * 32 + rp;
      float xs[4] = {areg[p].x, areg[p].y, areg[p].z, areg[p].w};
#pragma unroll
      for (int e = 0; e < 4; ++e)
        split_store(xs[e], &sHiA[r * LDSK + kk + e], &sLoA[r * LDSK + kk + e]);
    }
#pragma unroll
    for (int p = 0; p < 16; ++p) {
      const int kc = p * 2 + kt;
      split_store(breg[p], &sHiB[nn * LDSK + kc], &sLoB[nn * LDSK + kc]);
    }
    __syncthreads();

    // issue next k-step's global loads; latency hidden behind the WMMA block
    if (ks + 1 < SEQ / 32) {
      const int k0n = (ks + 1) * 32;
#pragma unroll
      for (int p = 0; p < 4; ++p)
        areg[p] = *reinterpret_cast<const float4*>(wbuf + (size_t)(mBase + p * 32 + rp) * SEQ + k0n + kk);
#pragma unroll
      for (int p = 0; p < 16; ++p)
        breg[p] = vb[(size_t)(k0n + p * 2 + kt) * EMB + nBase + nn];
    }

    v16h aHi[4], aLo[4], bHi[2], bLo[2];
#pragma unroll
    for (int mi = 0; mi < 4; ++mi) {
      const int base = (wm + mi * 16 + mr) * LDSK;
      aHi[mi] = frag_from(&sHiA[base + c0], &sHiA[base + c0 + 16]);
      aLo[mi] = frag_from(&sLoA[base + c0], &sLoA[base + c0 + 16]);
    }
#pragma unroll
    for (int ni = 0; ni < 2; ++ni) {
      const int base = (wn + ni * 16 + mr) * LDSK;
      bHi[ni] = frag_from(&sHiB[base + c0], &sHiB[base + c0 + 16]);
      bLo[ni] = frag_from(&sLoB[base + c0], &sLoB[base + c0 + 16]);
    }
#pragma unroll
    for (int mi = 0; mi < 4; ++mi)
#pragma unroll
      for (int ni = 0; ni < 2; ++ni) {
        const int t = mi * 2 + ni;
        acc[t] = __builtin_amdgcn_wmma_f32_16x16x32_f16(false, aHi[mi], false, bHi[ni], (short)0, acc[t], false, false);
        acc[t] = __builtin_amdgcn_wmma_f32_16x16x32_f16(false, aHi[mi], false, bLo[ni], (short)0, acc[t], false, false);
        acc[t] = __builtin_amdgcn_wmma_f32_16x16x32_f16(false, aLo[mi], false, bHi[ni], (short)0, acc[t], false, false);
      }
    __syncthreads();
  }

#pragma unroll
  for (int mi = 0; mi < 4; ++mi)
#pragma unroll
    for (int ni = 0; ni < 2; ++ni) {
      const int t = mi * 2 + ni;
      const int n = nBase + wn + ni * 16 + mr;
#pragma unroll
      for (int r = 0; r < 8; ++r) {
        const int m = mBase + wm + mi * 16 + hsel * 8 + r;
        ob[(size_t)m * EMB + n] = acc[t][r];
      }
    }
}

// ---------------------------------------------------------------------------
extern "C" void kernel_launch(void* const* d_in, const int* in_sizes, int n_in,
                              void* d_out, int out_size, void* d_ws, size_t ws_size,
                              hipStream_t stream) {
  const float* q = (const float*)d_in[0];
  const float* k = (const float*)d_in[1];
  const float* v = (const float*)d_in[2];
  float* out = (float*)d_out;

  const size_t O_ELEMS = (size_t)BATCH * SEQ * EMB;   // 16,777,216
  float* weights = out + O_ELEMS;                     // second output section

  // inverse-norm scratch: 2*BATCH*SEQ floats. Prefer d_ws; fall back to the
  // O-section of d_out (only written by the final kernel, stream-ordered safe).
  const size_t NORM_BYTES = (size_t)2 * BATCH * SEQ * sizeof(float);
  float* norms = (ws_size >= NORM_BYTES) ? (float*)d_ws : out;

  // 1) inverse norms: 2*BATCH*SEQ rows, 8 waves/block
  norms_kernel<<<(2 * BATCH * SEQ) / 8, 256, 0, stream>>>(q, k, norms);

  // 2) S = Qn @ Kn^T : grid (SEQ/128)^2 x BATCH
  qk_gemm_kernel<<<dim3(SEQ / 128, SEQ / 128, BATCH), 256, 0, stream>>>(q, k, norms, weights);

  // 3) O = S @ V : grid (EMB/128, SEQ/128, BATCH)
  sv_gemm_kernel<<<dim3(EMB / 128, SEQ / 128, BATCH), 256, 0, stream>>>(weights, v, out);
}
